// Encoder_63316407878007
// MI455X (gfx1250) — compile-verified, matrix operand order checked by reading, not run on hardware
//
#include <hip/hip_runtime.h>
#include <hip/hip_bf16.h>
#include <stdint.h>

typedef __bf16 bfloat;
typedef __attribute__((ext_vector_type(16))) __bf16 v16bf;
typedef __attribute__((ext_vector_type(8)))  __bf16 v8bf;
typedef __attribute__((ext_vector_type(8)))  float  v8f;
typedef __attribute__((ext_vector_type(4))) unsigned int u32x4;
typedef __attribute__((ext_vector_type(8))) int i32x8;
typedef __attribute__((ext_vector_type(4))) int i32x4;

#define WMMA_BF16(A, B, C) \
  __builtin_amdgcn_wmma_f32_16x16x32_bf16(false, (A), false, (B), (short)0, (C), false, false)

// ---------------------------------------------------------------------------
// Fragment loader for 16x16x32 bf16 WMMA (ISA 7.12.2 16-bit A layout).
// Works for A (row-major MxK) and for B when the weight is (N,K) row-major:
// lane (r = lane&15, half = lane>>4) holds two contiguous 8-element K-chunks
// at K offsets 8*half and 16+8*half of row r.
// ---------------------------------------------------------------------------
__device__ inline v16bf load_frag(const bfloat* __restrict__ base, int ld, int row0, int k0) {
  const int lane = threadIdx.x & 31;
  const int r    = lane & 15;
  const int half = lane >> 4;
  const bfloat* p = base + (size_t)(row0 + r) * ld + k0 + half * 8;
  v8bf c0 = *(const v8bf*)(p);
  v8bf c1 = *(const v8bf*)(p + 16);
  v16bf a;
#pragma unroll
  for (int i = 0; i < 8; ++i) { a[i] = c0[i]; a[i + 8] = c1[i]; }
  return a;
}

// gfx1250 hardware tanh (TRANS op). v_nops cover the TRANS result hazard,
// which the compiler cannot see through inline asm.
__device__ inline float tanh_hw(float x) {
  float r;
  asm volatile("v_tanh_f32 %0, %1\n\tv_nop\n\tv_nop" : "=v"(r) : "v"(x));
  return r;
}
__device__ inline float sigmoid_hw(float x) {
  return 0.5f * tanh_hw(0.5f * x) + 0.5f;
}

// ---------------------------------------------------------------------------
// TDM: DMA an 8-row x 1024-col f32 tile of xg (row stride 1024*1024 elements)
// from global memory into LDS.  Issued by one wave; tracked by TENSORcnt.
// D# per ISA ch.8: group0 = {count=1 | lds_addr | global_addr | type=2},
// group1 = {data_size=4B | tensor dims | tile dims | dim0 stride}.
// ---------------------------------------------------------------------------
__device__ inline void tdm_load_xg_tile(uint32_t lds_off, const float* gptr) {
  uint64_t ga = (uint64_t)(uintptr_t)gptr;
  u32x4 g0;
  g0[0] = 1u;                                                // count=1, user D#
  g0[1] = lds_off;                                           // lds_addr [63:32]
  g0[2] = (uint32_t)(ga & 0xffffffffu);                      // global_addr lo
  g0[3] = (uint32_t)((ga >> 32) & 0x01ffffffu) | (2u << 30); // ga[56:32] | type=2
  i32x8 g1;
  g1[0] = (int)(2u << 16);       // data_size = 2 (4 bytes)
  g1[1] = (int)(1024u << 16);    // tensor_dim0[15:0] in bits [31:16]
  g1[2] = (int)(8u << 16);       // tensor_dim1[15:0] in bits [31:16]
  g1[3] = (int)(1024u << 16);    // tile_dim0 in bits [127:112]
  g1[4] = 8;                     // tile_dim1
  g1[5] = (int)(1024u * 1024u);  // tensor_dim0_stride[31:0] (elements)
  g1[6] = 0;
  g1[7] = 0;
  i32x4 z4 = {0, 0, 0, 0};
#if __clang_major__ >= 23
  i32x8 z8 = {0, 0, 0, 0, 0, 0, 0, 0};
  __builtin_amdgcn_tensor_load_to_lds(g0, g1, z4, z4, z8, 0);
#else
  __builtin_amdgcn_tensor_load_to_lds(g0, g1, z4, z4, 0);
#endif
}

// ------------------------- elementwise converters --------------------------
__global__ void k_f2bf(const float* __restrict__ src, bfloat* __restrict__ dst, int n) {
  int i = blockIdx.x * blockDim.x + threadIdx.x;
  if (i < n) dst[i] = (bfloat)src[i];
}

__global__ void k_bias(const float* __restrict__ bihf, const float* __restrict__ bhhf,
                       const float* __restrict__ bihb, const float* __restrict__ bhhb,
                       float* __restrict__ out) {
  int i = blockIdx.x * blockDim.x + threadIdx.x;   // 0..2047
  if (i < 1024)      out[i] = bihf[i] + bhhf[i];
  else if (i < 2048) out[i] = bihb[i - 1024] + bhhb[i - 1024];
}

// ------------------- xg = x @ Wih^T + bias  (both dirs) --------------------
__global__ __launch_bounds__(256)
void k_xg(const bfloat* __restrict__ xb, const bfloat* __restrict__ wf,
          const bfloat* __restrict__ wb, const float* __restrict__ bias,
          float* __restrict__ xg) {
  const int wave = (blockIdx.x * blockDim.x + threadIdx.x) >> 5;   // 0..16383
  const int mrow = (wave >> 5) << 4;       // 512 row strips of 16
  const int ncol = (wave & 31) << 6;       // 32 col strips of 64 over N=2048
  const int dir  = ncol >> 10;
  const int nl0  = ncol & 1023;
  const bfloat* W = dir ? wb : wf;
  v8f acc[4] = {};
  for (int kk = 0; kk < 768; kk += 32) {
    v16bf a = load_frag(xb, 768, mrow, kk);
#pragma unroll
    for (int t = 0; t < 4; ++t) {
      v16bf b = load_frag(W, 768, nl0 + 16 * t, kk);
      acc[t] = WMMA_BF16(a, b, acc[t]);
    }
  }
  const int lane = threadIdx.x & 31, nl = lane & 15, half = lane >> 4;
  float* out = xg + (size_t)dir * (8192u * 1024u);
#pragma unroll
  for (int t = 0; t < 4; ++t) {
    int n = nl0 + 16 * t + nl;
    float bv = bias[dir * 1024 + n];
#pragma unroll
    for (int j = 0; j < 8; ++j) {
      int m = mrow + j + 8 * half;
      out[(size_t)m * 1024 + n] = acc[t][j] + bv;
    }
  }
}

// ----------------------------- LSTM recurrence -----------------------------
// One block (32 waves) per direction. h (8x256 padded to 16 rows) in LDS as
// bf16 (WMMA A side); c in registers; xg[t] tiles double-buffered in LDS via
// TDM prefetch of step t+1 overlapped with step t's WMMA + gate work.
__global__ __launch_bounds__(1024)
void k_lstm(const bfloat* __restrict__ whhf, const bfloat* __restrict__ whhb,
            const float* __restrict__ xg, float* __restrict__ hcat,
            bfloat* __restrict__ hcat_bf, bfloat* __restrict__ hT_bf) {
  __shared__ alignas(16) bfloat h_lds[16 * 256];
  __shared__ float g_lds[8 * 1024];
  __shared__ alignas(16) float xg_lds[2][8 * 1024];
  const int tid = threadIdx.x;
  const int dir = blockIdx.x;
  const bfloat* W = dir ? whhb : whhf;
  const float* xgd = xg + (size_t)dir * (8192u * 1024u);
  const uint32_t xg_off[2] = {(uint32_t)(uintptr_t)(void*)&xg_lds[0][0],
                              (uint32_t)(uintptr_t)(void*)&xg_lds[1][0]};

  for (int i = tid; i < 16 * 256; i += 1024) h_lds[i] = (bfloat)0.0f;
  const int m0 = tid >> 8, j0 = tid & 255;    // cell A: batches 0..3
  const int m1 = m0 + 4;                       // cell B: batches 4..7
  float c0 = 0.0f, c1 = 0.0f;
  const int wave = tid >> 5;
  const int n0 = wave << 5;                    // 32-col strip per wave
  const int lane = tid & 31, nl = lane & 15, half = lane >> 4;

  if (tid < 32) {                              // wave 0: preload xg[step 0]
    tdm_load_xg_tile(xg_off[0], xgd + (size_t)(dir ? 1023 : 0) * 1024);
    __builtin_amdgcn_s_wait_tensorcnt(0);
  }
  __syncthreads();

  for (int step = 0; step < 1024; ++step) {
    const int t = dir ? (1023 - step) : step;
    if (step + 1 < 1024 && tid < 32) {         // async prefetch of next tile
      int tn = dir ? (1023 - (step + 1)) : (step + 1);
      tdm_load_xg_tile(xg_off[(step + 1) & 1], xgd + (size_t)tn * 1024);
    }
    v8f acc0 = {}, acc1 = {};
#pragma unroll
    for (int kk = 0; kk < 256; kk += 32) {
      v16bf a  = load_frag(h_lds, 256, 0, kk);
      v16bf b0 = load_frag(W, 256, n0, kk);
      v16bf b1 = load_frag(W, 256, n0 + 16, kk);
      acc0 = WMMA_BF16(a, b0, acc0);
      acc1 = WMMA_BF16(a, b1, acc1);
    }
    const float* xx = xg_lds[step & 1];
    if (half == 0) {   // rows 8..15 of the padded tile are zero/unused
#pragma unroll
      for (int j = 0; j < 8; ++j) {
        g_lds[j * 1024 + n0 + nl]      = acc0[j] + xx[j * 1024 + n0 + nl];
        g_lds[j * 1024 + n0 + 16 + nl] = acc1[j] + xx[j * 1024 + n0 + 16 + nl];
      }
    }
    __syncthreads();   // g ready; all h_lds / xg-buffer reads done
    {
      const float* gm = g_lds + m0 * 1024;
      float ig = sigmoid_hw(gm[j0]);
      float fg = sigmoid_hw(gm[256 + j0]);
      float gg = tanh_hw(gm[512 + j0]);
      float og = sigmoid_hw(gm[768 + j0]);
      c0 = fg * c0 + ig * gg;
      float h = og * tanh_hw(c0);
      h_lds[m0 * 256 + j0] = (bfloat)h;
      size_t oi = ((size_t)m0 * 1024 + t) * 512 + dir * 256 + j0;
      hcat[oi] = h;  hcat_bf[oi] = (bfloat)h;
      hT_bf[((size_t)m0 * 512 + dir * 256 + j0) * 1024 + t] = (bfloat)h;
    }
    {
      const float* gm = g_lds + m1 * 1024;
      float ig = sigmoid_hw(gm[j0]);
      float fg = sigmoid_hw(gm[256 + j0]);
      float gg = tanh_hw(gm[512 + j0]);
      float og = sigmoid_hw(gm[768 + j0]);
      c1 = fg * c1 + ig * gg;
      float h = og * tanh_hw(c1);
      h_lds[m1 * 256 + j0] = (bfloat)h;
      size_t oi = ((size_t)m1 * 1024 + t) * 512 + dir * 256 + j0;
      hcat[oi] = h;  hcat_bf[oi] = (bfloat)h;
      hT_bf[((size_t)m1 * 512 + dir * 256 + j0) * 1024 + t] = (bfloat)h;
    }
    if (tid < 32) __builtin_amdgcn_s_wait_tensorcnt(0);  // next tile landed
    __syncthreads();   // h_lds + next xg buffer ready for next step
  }
}

// ----------------------- qp/kp projections (X=10) --------------------------
__global__ void k_qk(const float* __restrict__ h, const float* __restrict__ W1,
                     float* __restrict__ qp, float* __restrict__ kp) {
  const int row = blockIdx.x;           // 0..8191
  const int tid = threadIdx.x;          // 32 threads, 20 active
  if (tid >= 20) return;
  const int x = tid % 10;
  const int isk = tid / 10;
  const float* hr = h + (size_t)row * 512;
  const float* w  = W1 + x * 1024 + isk * 512;
  float s = 0.0f;
  for (int k = 0; k < 512; ++k) s += hr[k] * w[k];
  (isk ? kp : qp)[(size_t)row * 10 + x] = s;
}

// ---------------- scores + masked softmax -> attn (bf16) -------------------
__global__ __launch_bounds__(256)
void k_attn(const float* __restrict__ qp, const float* __restrict__ kp,
            const float* __restrict__ W2, const int* __restrict__ mask,
            bfloat* __restrict__ attn) {
  const int bt = blockIdx.x;            // b*1024 + t
  const int b  = bt >> 10;
  const int tid = threadIdx.x;
  __shared__ float red[256];
  __shared__ float qv[10], w2v[10];
  if (tid < 10) { qv[tid] = qp[(size_t)bt * 10 + tid]; w2v[tid] = W2[tid]; }
  __syncthreads();
  float s[4];
#pragma unroll
  for (int it = 0; it < 4; ++it) {
    int u = tid + 256 * it;
    const float* kpu = kp + ((size_t)b * 1024 + u) * 10;
    float a = 0.0f;
#pragma unroll
    for (int x = 0; x < 10; ++x) a += tanh_hw(qv[x] + kpu[x]) * w2v[x];
    s[it] = (mask[b * 1024 + u] == 0) ? -1e30f : a;
  }
  float mx = fmaxf(fmaxf(s[0], s[1]), fmaxf(s[2], s[3]));
  red[tid] = mx; __syncthreads();
  for (int o = 128; o > 0; o >>= 1) { if (tid < o) red[tid] = fmaxf(red[tid], red[tid + o]); __syncthreads(); }
  mx = red[0]; __syncthreads();
  float e[4]; float sum = 0.0f;
#pragma unroll
  for (int it = 0; it < 4; ++it) { e[it] = __expf(s[it] - mx); sum += e[it]; }
  red[tid] = sum; __syncthreads();
  for (int o = 128; o > 0; o >>= 1) { if (tid < o) red[tid] += red[tid + o]; __syncthreads(); }
  float inv = 1.0f / (red[0] + 1e-20f);
#pragma unroll
  for (int it = 0; it < 4; ++it)
    attn[(size_t)bt * 1024 + tid + 256 * it] = (bfloat)(e[it] * inv);
}

// --------------- context[b] = attn[b] @ h[b]  (per-batch GEMM) -------------
__global__ __launch_bounds__(256)
void k_ctx(const bfloat* __restrict__ attn, const bfloat* __restrict__ hT,
           bfloat* __restrict__ ctx) {
  const int wave = (blockIdx.x * blockDim.x + threadIdx.x) >> 5;  // 0..4095
  const int bidx = wave >> 9;            // 512 waves per batch
  const int wl   = wave & 511;
  const int mrow = (wl >> 3) << 4;       // 64 strips of 16 (T rows)
  const int ncol = (wl & 7) << 6;        // 8 strips of 64 (D cols)
  const bfloat* A  = attn + (size_t)bidx * 1024 * 1024;
  const bfloat* Bm = hT   + (size_t)bidx * 512 * 1024;   // (d,u) row-major
  v8f acc[4] = {};
  for (int kk = 0; kk < 1024; kk += 32) {
    v16bf a = load_frag(A, 1024, mrow, kk);
#pragma unroll
    for (int t = 0; t < 4; ++t)
      acc[t] = WMMA_BF16(a, load_frag(Bm, 1024, ncol + 16 * t, kk), acc[t]);
  }
  const int lane = threadIdx.x & 31, nl = lane & 15, half = lane >> 4;
  bfloat* out = ctx + (size_t)bidx * 1024 * 512;
#pragma unroll
  for (int t = 0; t < 4; ++t)
#pragma unroll
    for (int j = 0; j < 8; ++j)
      out[(size_t)(mrow + j + 8 * half) * 512 + ncol + 16 * t + nl] = (bfloat)acc[t][j];
}

// ------------- y = [h, ctx] @ W3^T + b3   M=8192 N=512 K=1024 --------------
__global__ __launch_bounds__(256)
void k_out(const bfloat* __restrict__ hbf, const bfloat* __restrict__ ctx,
           const bfloat* __restrict__ w3, const float* __restrict__ b3,
           float* __restrict__ y) {
  const int wave = (blockIdx.x * blockDim.x + threadIdx.x) >> 5;  // 0..4095
  const int mrow = (wave >> 3) << 4;     // 512 strips of 16
  const int ncol = (wave & 7) << 6;      // 8 strips of 64
  v8f acc[4] = {};
  for (int kk = 0; kk < 1024; kk += 32) {
    const bfloat* A = (kk < 512) ? hbf : ctx;
    int ka = (kk < 512) ? kk : kk - 512;
    v16bf a = load_frag(A, 512, mrow, ka);
#pragma unroll
    for (int t = 0; t < 4; ++t)
      acc[t] = WMMA_BF16(a, load_frag(w3, 1024, ncol + 16 * t, kk), acc[t]);
  }
  const int lane = threadIdx.x & 31, nl = lane & 15, half = lane >> 4;
#pragma unroll
  for (int t = 0; t < 4; ++t) {
    int n = ncol + 16 * t + nl;
    float bv = b3[n];
#pragma unroll
    for (int j = 0; j < 8; ++j)
      y[(size_t)(mrow + j + 8 * half) * 512 + n] = acc[t][j] + bv;
  }
}

// ---------------------------------------------------------------------------
extern "C" void kernel_launch(void* const* d_in, const int* in_sizes, int n_in,
                              void* d_out, int out_size, void* d_ws, size_t ws_size,
                              hipStream_t stream) {
  const float* x     = (const float*)d_in[0];
  const int*   mask  = (const int*)d_in[1];
  const float* Wih_f = (const float*)d_in[3];
  const float* Whh_f = (const float*)d_in[4];
  const float* bih_f = (const float*)d_in[5];
  const float* bhh_f = (const float*)d_in[6];
  const float* Wih_b = (const float*)d_in[7];
  const float* Whh_b = (const float*)d_in[8];
  const float* bih_b = (const float*)d_in[9];
  const float* bhh_b = (const float*)d_in[10];
  const float* W1    = (const float*)d_in[11];
  const float* W2    = (const float*)d_in[12];
  const float* W3    = (const float*)d_in[13];
  const float* b3    = (const float*)d_in[14];
  float* y = (float*)d_out;

  char* p = (char*)d_ws;
  auto carve = [&](size_t bytes) -> char* {
    char* r = p;
    p += (bytes + 255) & ~(size_t)255;
    return r;
  };
  bfloat* x_bf    = (bfloat*)carve(8ull * 1024 * 768 * 2);
  bfloat* wf_bf   = (bfloat*)carve(1024ull * 768 * 2);
  bfloat* wb_bf   = (bfloat*)carve(1024ull * 768 * 2);
  bfloat* whhf_bf = (bfloat*)carve(1024ull * 256 * 2);
  bfloat* whhb_bf = (bfloat*)carve(1024ull * 256 * 2);
  bfloat* w3_bf   = (bfloat*)carve(512ull * 1024 * 2);
  float*  bias    = (float*) carve(2048ull * 4);
  float*  xg      = (float*) carve(2ull * 8192 * 1024 * 4);
  float*  hcat    = (float*) carve(8192ull * 512 * 4);
  bfloat* hcat_bf = (bfloat*)carve(8192ull * 512 * 2);
  bfloat* hT_bf   = (bfloat*)carve(8ull * 512 * 1024 * 2);
  float*  qp      = (float*) carve(8192ull * 10 * 4);
  float*  kp      = (float*) carve(8192ull * 10 * 4);
  bfloat* attn    = (bfloat*)carve(8ull * 1024 * 1024 * 2);
  bfloat* ctx     = (bfloat*)carve(8192ull * 512 * 2);

  // 1) bf16 conversions of activations / weights
  k_f2bf<<<(6291456 + 255) / 256, 256, 0, stream>>>(x, x_bf, 6291456);
  k_f2bf<<<(786432 + 255) / 256, 256, 0, stream>>>(Wih_f, wf_bf, 786432);
  k_f2bf<<<(786432 + 255) / 256, 256, 0, stream>>>(Wih_b, wb_bf, 786432);
  k_f2bf<<<(262144 + 255) / 256, 256, 0, stream>>>(Whh_f, whhf_bf, 262144);
  k_f2bf<<<(262144 + 255) / 256, 256, 0, stream>>>(Whh_b, whhb_bf, 262144);
  k_f2bf<<<(524288 + 255) / 256, 256, 0, stream>>>(W3, w3_bf, 524288);
  k_bias<<<8, 256, 0, stream>>>(bih_f, bhh_f, bih_b, bhh_b, bias);

  // 2) input projection GEMM (both directions), f32 out
  k_xg<<<2048, 256, 0, stream>>>(x_bf, wf_bf, wb_bf, bias, xg);

  // 3) sequential BiLSTM recurrence: 1 block per direction (TDM-prefetched xg)
  k_lstm<<<2, 1024, 0, stream>>>(whhf_bf, whhb_bf, xg, hcat, hcat_bf, hT_bf);

  // 4) attention projections, scores+softmax, context GEMM
  k_qk<<<8192, 32, 0, stream>>>(hcat, W1, qp, kp);
  k_attn<<<8192, 256, 0, stream>>>(qp, kp, W2, mask, attn);
  k_ctx<<<512, 256, 0, stream>>>(attn, hT_bf, ctx);

  // 5) output projection
  k_out<<<512, 256, 0, stream>>>(hcat_bf, ctx, w3_bf, b3, y);
}
